// WaveFieldAttention_84043920048464
// MI455X (gfx1250) — matrix operand order repaired and checked by reading, not compiled, not run on hardware
//
#include <hip/hip_runtime.h>
#include <hip/hip_bf16.h>
#include <math.h>

// WaveFieldAttention on gfx1250 (MI455X).
// FFT conv -> precomputed per-head N x N causal operator A_h, so the whole
// module becomes GEMMs on v_wmma_f32_16x16x32_f16 (f16 in, f32 accumulate).
// GEMM: block tile 128x64, 8 waves, each wave 32x32 (4 wmma / K-step),
// float4 global loads, packed v4h LDS stores, WMMA-swizzled LDS layout,
// double-buffered LDS (one barrier per K-step, loads overlap WMMA).

typedef _Float16 half_t;
typedef __attribute__((ext_vector_type(4)))  _Float16 v4h;
typedef __attribute__((ext_vector_type(16))) _Float16 v16h;
typedef __attribute__((ext_vector_type(8)))  float    v8f;

#define Bsz   8
#define Ntok  1024
#define EMB   1024
#define Hh    16
#define HD    64
#define Gg    4096

// ---------------------------------------------------------------- kern build
__global__ void build_kern(const float* __restrict__ freq,
                           const float* __restrict__ damp,
                           const float* __restrict__ phase,
                           float* __restrict__ kern) {
  int h = blockIdx.x;
  int t0 = threadIdx.x;
  float alpha = log1pf(expf(damp[h]));     // softplus
  float fq = freq[h], ph = phase[h];
  float* kh = kern + h * Gg;
  float lsum = 0.f;
  for (int t = t0; t < Gg; t += 256) {
    float v = expf(-alpha * (float)t) * cosf(fq * (float)t + ph);
    kh[t] = v;
    lsum += fabsf(v);
  }
  __shared__ float red[256];
  red[t0] = lsum;
  __syncthreads();
  for (int s = 128; s > 0; s >>= 1) {
    if (t0 < s) red[t0] += red[t0 + s];
    __syncthreads();
  }
  float inv = 1.f / fmaxf(red[0], 1e-8f);
  for (int t = t0; t < Gg; t += 256) kh[t] *= inv;
}

// ------------------------------------------------------------ coupling softmax
__global__ void coupling_kernel(const float* __restrict__ fc, float* __restrict__ coup) {
  int r = threadIdx.x;
  if (r < Hh) {
    float mx = -1e30f;
    for (int j = 0; j < Hh; ++j) mx = fmaxf(mx, fc[r * Hh + j]);
    float e[Hh]; float s = 0.f;
    for (int j = 0; j < Hh; ++j) { e[j] = expf(fc[r * Hh + j] - mx); s += e[j]; }
    for (int j = 0; j < Hh; ++j) coup[r * Hh + j] = e[j] / s;
  }
}

// ------------------------------------------- per-head gather/conv operator A_h
__device__ inline void pos_of(int n, int& lo, float& fr) {
  double p = (double)n * (4095.0 / 1023.0);
  if (p > 4094.0) p = 4094.0;
  int l = (int)p; if (l > 4094) l = 4094; if (l < 0) l = 0;
  double f = p - (double)l; if (f < 0.0) f = 0.0; if (f > 1.0) f = 1.0;
  lo = l; fr = (float)f;
}

__global__ void build_A(const float* __restrict__ kern, float* __restrict__ Amat) {
  int h = blockIdx.z;
  int m = blockIdx.y * 16 + threadIdx.y;
  int n = blockIdx.x * 16 + threadIdx.x;
  int lom; float fm; pos_of(m, lom, fm);
  int lon; float fn; pos_of(n, lon, fn);
  const float* kh = kern + h * Gg;
  float wlm = 1.f - fm, whm = fm, wln = 1.f - fn, whn = fn;
  int him = lom + 1, hin = lon + 1;
  float acc = 0.f; int d;
  d = lom - lon; if (d >= 0) acc += wlm * wln * kh[d];
  d = lom - hin; if (d >= 0) acc += wlm * whn * kh[d];
  d = him - lon; if (d >= 0) acc += whm * wln * kh[d];
  d = him - hin; if (d >= 0) acc += whm * whn * kh[d];
  Amat[((size_t)h * Ntok + m) * Ntok + n] = acc;
}

// --------------------------------------------------------------- unified GEMM
// C[M x Nc] = act( X[M x K] @ W[Nc x K]^T + bias ), all row-major fp32.
// batch != 0: blockIdx.z = b*H + h; X <- Amat + h*N*N, W <- depT + z*HD*N,
//             C <- G1 + z*N*HD.
// Block tile 128(M) x 64(N), Kc = 32. 8 waves: wm = wave>>1 (0..3),
// wn = wave&1 (0..1); each wave computes 32x32 via 4 WMMA per K-step.
// Double-buffered LDS: iteration i reads buf i&1, writes buf (i+1)&1;
// one barrier per K-step, next-tile global loads issued before the WMMAs.
__global__ __launch_bounds__(256) void gemm_wmma(const float* __restrict__ X,
                                                 const float* __restrict__ W,
                                                 const float* __restrict__ bias,
                                                 float* __restrict__ C,
                                                 int M, int Nc, int K,
                                                 int act, int batch) {
  if (batch) {
    int z = blockIdx.z;
    int h = z & (Hh - 1);
    X += (size_t)h << 20;            // h * 1024 * 1024
    W += (size_t)z * HD * Ntok;      // depT slab (HD rows x Ntok)
    C += (size_t)z * Ntok * HD;
  }
  __shared__ v16h As[2][8][32];   // 2 x 8 KB
  __shared__ v16h Bs[2][4][32];   // 2 x 4 KB
  int tid  = threadIdx.x;
  int lane = tid & 31, wave = tid >> 5;
  int wm = wave >> 1, wn = wave & 1;
  int row0 = blockIdx.y * 128;
  int col0 = blockIdx.x * 64;
  v8f acc00 = {}, acc01 = {}, acc10 = {}, acc11 = {};

  float4 ra[4], rb[2];
  auto loadGlobal = [&](int k0) {
#pragma unroll
    for (int i = 0; i < 4; ++i) {
      int flat4 = tid + i * 256;
      int m  = flat4 >> 3;
      int k4 = (flat4 & 7) << 2;
      ra[i] = *(const float4*)&X[(size_t)(row0 + m) * K + k0 + k4];
    }
#pragma unroll
    for (int i = 0; i < 2; ++i) {
      int flat4 = tid + i * 256;
      int n  = flat4 >> 3;
      int k4 = (flat4 & 7) << 2;
      rb[i] = *(const float4*)&W[(size_t)(col0 + n) * K + k0 + k4];
    }
  };
  // Swizzle into WMMA operand layout.
  // A: lane l holds row M=l%16; halves j 0..7 = K (l/16)*8..+7, 8..15 = +16.
  // B: lane l holds col N=l%16; same K mapping as A.
  // 4 consecutive K -> 4 consecutive j slots -> one v4h (ds_store_b64).
  auto storeLds = [&](int buf) {
    half_t* Ah = (half_t*)As[buf];
    half_t* Bh = (half_t*)Bs[buf];
#pragma unroll
    for (int i = 0; i < 4; ++i) {
      int flat4 = tid + i * 256;
      int m  = flat4 >> 3;
      int k4 = (flat4 & 7) << 2;
      int lidx = (((k4 & 15) >> 3) << 4) | (m & 15);
      int j0   = ((k4 >> 4) << 3) | (k4 & 7);
      v4h p = {(half_t)ra[i].x, (half_t)ra[i].y, (half_t)ra[i].z, (half_t)ra[i].w};
      *(v4h*)(Ah + (((m >> 4) * 32 + lidx) << 4) + j0) = p;
    }
#pragma unroll
    for (int i = 0; i < 2; ++i) {
      int flat4 = tid + i * 256;
      int n  = flat4 >> 3;
      int k4 = (flat4 & 7) << 2;
      int lidx = (((k4 & 15) >> 3) << 4) | (n & 15);
      int j0   = ((k4 >> 4) << 3) | (k4 & 7);
      v4h p = {(half_t)rb[i].x, (half_t)rb[i].y, (half_t)rb[i].z, (half_t)rb[i].w};
      *(v4h*)(Bh + (((n >> 4) * 32 + lidx) << 4) + j0) = p;
    }
  };

  int nIter = K >> 5;
  loadGlobal(0);
  storeLds(0);
  __syncthreads();
  for (int it = 0; it < nIter; ++it) {
    int cur = it & 1;
    bool more = (it + 1) < nIter;           // uniform across block
    if (more) loadGlobal((it + 1) << 5);    // overlap with WMMA below
    v16h a0 = As[cur][wm * 2][lane];
    v16h a1 = As[cur][wm * 2 + 1][lane];
    v16h b0 = Bs[cur][wn * 2][lane];
    v16h b1 = Bs[cur][wn * 2 + 1][lane];
    acc00 = __builtin_amdgcn_wmma_f32_16x16x32_f16(false, a0, false, b0, (short)0, acc00, false, false);
    acc01 = __builtin_amdgcn_wmma_f32_16x16x32_f16(false, a0, false, b1, (short)0, acc01, false, false);
    acc10 = __builtin_amdgcn_wmma_f32_16x16x32_f16(false, a1, false, b0, (short)0, acc10, false, false);
    acc11 = __builtin_amdgcn_wmma_f32_16x16x32_f16(false, a1, false, b1, (short)0, acc11, false, false);
    if (more) storeLds(cur ^ 1);
    __syncthreads();
  }

  // ---- epilogue.  D layout: c[r] = D[r + 8*(lane/16), lane%16]
  int cb = col0 + wn * 32 + (lane & 15);
  int rb_ = row0 + wm * 32 + ((lane >> 4) << 3);
#pragma unroll
  for (int mi = 0; mi < 2; ++mi) {
#pragma unroll
    for (int ni = 0; ni < 2; ++ni) {
      const v8f& a = mi == 0 ? (ni == 0 ? acc00 : acc01) : (ni == 0 ? acc10 : acc11);
      int col = cb + ni * 16;
      float bv = bias ? bias[col] : 0.f;
      int rowb = rb_ + mi * 16;
#pragma unroll
      for (int r = 0; r < 8; ++r) {
        float v = a[r] + bv;
        if (act == 1) v = 1.f / (1.f + expf(-v));
        C[(size_t)(rowb + r) * Nc + col] = v;
      }
    }
  }
}

// --------------------------------------------------------- k magnitude pass
// kv layout: [b*N + n][2048], cols 0..1023 = k, 1024..2047 = v
__global__ void kmag_kernel(const float* __restrict__ kv, float* __restrict__ kmag) {
  int idx = blockIdx.x * blockDim.x + threadIdx.x;   // (b*H + h)*N + n
  int n  = idx & (Ntok - 1);
  int bh = idx >> 10;
  int h  = bh & (Hh - 1);
  int b  = bh >> 4;
  const float* krow = kv + ((size_t)(b * Ntok + n)) * 2048 + h * HD;
  float s = 0.f;
  for (int d = 0; d < HD; ++d) { float kk = krow[d]; s += kk * kk; }
  kmag[idx] = sqrtf(s);
}

// -------------------------------- deposit, transposed: depT[b,h][d][n_token]
__global__ void depT_kernel(const float* __restrict__ kv, const float* __restrict__ kmag,
                            float* __restrict__ depT) {
  size_t idx = (size_t)blockIdx.x * 256 + threadIdx.x;  // ((b*H+h)*HD + d)*N + n
  int n    = (int)(idx & (Ntok - 1));
  size_t r = idx >> 10;
  int d    = (int)(r & (HD - 1));
  int bh   = (int)(r >> 6);
  int h    = bh & (Hh - 1);
  int b    = bh >> 4;
  float v = kv[((size_t)(b * Ntok + n)) * 2048 + EMB + h * HD + d];
  depT[idx] = v * kmag[(size_t)bh * Ntok + n];
}

// ------------------------------------- head-mix (coupling) * gate -> y (B,N,EMB)
__global__ void mix_kernel(const float* __restrict__ G1, const float* __restrict__ gate,
                           const float* __restrict__ coup, float* __restrict__ y) {
  __shared__ float c[Hh * Hh];
  if (threadIdx.x < Hh * Hh) c[threadIdx.x] = coup[threadIdx.x];
  __syncthreads();
  size_t idx = (size_t)blockIdx.x * 256 + threadIdx.x;   // over B*N*EMB
  int d = (int)(idx & (HD - 1));
  int h = (int)((idx >> 6) & (Hh - 1));
  size_t bn = idx >> 10;
  int n = (int)(bn & (Ntok - 1));
  int b = (int)(bn >> 10);
  const float* cr = c + h * Hh;
  float acc = 0.f;
  for (int k = 0; k < Hh; ++k)
    acc += cr[k] * G1[(((size_t)(b * Hh + k) * Ntok + n) << 6) + d];
  y[idx] = acc * gate[idx];
}

// ---------------------------------------------------------------- launch
extern "C" void kernel_launch(void* const* d_in, const int* in_sizes, int n_in,
                              void* d_out, int out_size, void* d_ws, size_t ws_size,
                              hipStream_t stream) {
  const float* x      = (const float*)d_in[0];
  const float* Wqkv   = (const float*)d_in[1];
  const float* bqkv   = (const float*)d_in[2];
  const float* Wout   = (const float*)d_in[3];
  const float* bout   = (const float*)d_in[4];
  const float* Wgate  = (const float*)d_in[5];
  const float* bgate  = (const float*)d_in[6];
  const float* wfreq  = (const float*)d_in[7];
  const float* wdamp  = (const float*)d_in[8];
  const float* wphase = (const float*)d_in[9];
  const float* fcoup  = (const float*)d_in[10];
  float* out = (float*)d_out;

  char* wsp = (char*)d_ws;
  auto alloc = [&](size_t bytes) {
    float* p = (float*)wsp;
    wsp += (bytes + 255) & ~(size_t)255;
    return p;
  };
  const size_t M = (size_t)Bsz * Ntok;                    // 8192
  float* kern = alloc((size_t)Hh * Gg * 4);               // 256 KB
  float* coup = alloc((size_t)Hh * Hh * 4);
  float* kmag = alloc((size_t)Bsz * Hh * Ntok * 4);       // 512 KB
  float* Amat = alloc((size_t)Hh * Ntok * Ntok * 4);      // 64 MB
  float* kv   = alloc(M * 2048 * 4);                      // 64 MB
  float* gate = alloc(M * EMB * 4);                       // 32 MB
  float* depT = alloc((size_t)Bsz * Hh * HD * Ntok * 4);  // 32 MB
  float* G1   = alloc((size_t)Bsz * Hh * Ntok * HD * 4);  // 32 MB
  float* y    = alloc(M * EMB * 4);                       // 32 MB

  build_kern<<<Hh, 256, 0, stream>>>(wfreq, wdamp, wphase, kern);
  coupling_kernel<<<1, 32, 0, stream>>>(fcoup, coup);
  build_A<<<dim3(Ntok / 16, Ntok / 16, Hh), dim3(16, 16), 0, stream>>>(kern, Amat);

  // [k;v] = x @ Wqkv[1024:3072].T   (q is dead in the reference)
  gemm_wmma<<<dim3(2048 / 64, M / 128, 1), 256, 0, stream>>>(
      x, Wqkv + (size_t)EMB * EMB, bqkv + EMB, kv, (int)M, 2048, EMB, 0, 0);
  // gate = sigmoid(x @ Wgate.T + bgate)
  gemm_wmma<<<dim3(EMB / 64, M / 128, 1), 256, 0, stream>>>(
      x, Wgate, bgate, gate, (int)M, EMB, EMB, 1, 0);

  kmag_kernel<<<(Bsz * Hh * Ntok) / 256, 256, 0, stream>>>(kv, kmag);
  depT_kernel<<<(unsigned)((size_t)Bsz * Hh * HD * Ntok / 256), 256, 0, stream>>>(
      kv, kmag, depT);

  // G1[b,h] = A_h @ depT[b,h]^T   (batched over z = b*H + h)
  gemm_wmma<<<dim3(1, Ntok / 128, Bsz * Hh), 256, 0, stream>>>(
      Amat, depT, nullptr, G1, Ntok, HD, Ntok, 0, 1);

  mix_kernel<<<(unsigned)((size_t)Bsz * Ntok * EMB / 256), 256, 0, stream>>>(
      G1, gate, coup, y);

  // out = y @ Wout.T + bout
  gemm_wmma<<<dim3(EMB / 64, M / 128, 1), 256, 0, stream>>>(
      y, Wout, bout, out, (int)M, EMB, EMB, 0, 0);
}
